// SelfAttention_13280038880163
// MI455X (gfx1250) — matrix operand order repaired and checked by reading, not compiled, not run on hardware
//
#include <hip/hip_runtime.h>
#include <stdint.h>

typedef __bf16 bf16_t;
typedef __attribute__((ext_vector_type(16))) __bf16 v16bf;
typedef __attribute__((ext_vector_type(8)))  __bf16 v8bf;
typedef __attribute__((ext_vector_type(8)))  float  v8f;

#define S_LEN  2048
#define NBATCH 2
#define NHEADS 16
#define DHEAD  64
#define EMB    1024

static __device__ __forceinline__ v8f vzero8() {
  v8f z;
#pragma unroll
  for (int i = 0; i < 8; ++i) z[i] = 0.0f;
  return z;
}

static __device__ __forceinline__ v16bf comb(v8bf lo, v8bf hi) {
  return __builtin_shufflevector(lo, hi, 0, 1, 2, 3, 4, 5, 6, 7,
                                         8, 9, 10, 11, 12, 13, 14, 15);
}

static __device__ __forceinline__ v8f wmma_bf16(v16bf a, v16bf b, v8f c) {
  // D = A(16x32 bf16) * B(32x16 bf16) + C(16x16 f32)
  return __builtin_amdgcn_wmma_f32_16x16x32_bf16(
      /*neg_a=*/false, a, /*neg_b=*/false, b,
      /*c_mod=*/(short)0, c, /*reuse_a=*/false, /*reuse_b=*/false);
}

// gfx1250 async global->LDS copy (ASYNCcnt-tracked, no VGPR data).
// GVS mode: mem = SGPR64 + VGPR32 + IOFFSET; LDS = VDST vgpr + IOFFSET.
static __device__ __forceinline__ void async_b128(uint32_t lds_off, uint32_t voff,
                                                  unsigned long long sbase) {
  asm volatile("global_load_async_to_lds_b128 %0, %1, %2 offset:0"
               :: "v"(lds_off), "v"(voff), "s"(sbase)
               : "memory");
}
static __device__ __forceinline__ void wait_async_le4() {
  asm volatile("s_wait_asynccnt 0x4" ::: "memory");
}
static __device__ __forceinline__ void wait_async_le0() {
  asm volatile("s_wait_asynccnt 0x0" ::: "memory");
}

// ---------------------------------------------------------------------------
// Kernel 1: per-head QKV projection. out[n,h,s,:] = in[n,s,h,:] @ W^T (bf16).
// Q is pre-scaled by 1/sqrt(E) so the attention kernel skips the energy scale.
// V is stored transposed: Vt[n,h,d,s] so PV B-fragments are contiguous later.
// ---------------------------------------------------------------------------
__global__ __launch_bounds__(256) void proj_kernel(
    const float* __restrict__ q_in, const float* __restrict__ k_in,
    const float* __restrict__ v_in, const float* __restrict__ Wq,
    const float* __restrict__ Wk, const float* __restrict__ Wv,
    bf16_t* __restrict__ Qp, bf16_t* __restrict__ Kp, bf16_t* __restrict__ Vt)
{
  const int which = blockIdx.z % 3;  // 0=Q, 1=K, 2=V
  const int n     = blockIdx.z / 3;
  const int h     = blockIdx.y;
  const int s0    = blockIdx.x * 128;
  const int tid   = threadIdx.x;
  const int wave  = tid >> 5, lane = tid & 31;
  const int lo = lane & 15, hi = lane >> 4;

  const float* in = (which == 0) ? q_in : (which == 1) ? k_in : v_in;
  const float* W  = (which == 0) ? Wq   : (which == 1) ? Wk   : Wv;
  const float osc = (which == 0) ? 0.03125f : 1.0f;  // fold 1/sqrt(1024) into Q

  __shared__ __align__(32) bf16_t wsm[DHEAD * DHEAD];  // W row-major (o,d) bf16
#pragma unroll 4
  for (int i = tid; i < DHEAD * DHEAD; i += 256) wsm[i] = (bf16_t)W[i];
  __syncthreads();

  // A fragments: row M = lane%16 within this wave's 16-row slice
  const int arow  = s0 + wave * 16 + lo;
  const float* rp = in + ((size_t)(n * S_LEN + arow) * NHEADS + h) * DHEAD;

  v16bf a[2];
#pragma unroll
  for (int ks = 0; ks < 2; ++ks) {
    const int kc = ks * 32 + hi * 16;  // contiguous k order (B matches): lanes<16 get K 0..15
    v16bf t;
#pragma unroll
    for (int j = 0; j < 8; ++j) t[j]     = (bf16_t)rp[kc + j];
#pragma unroll
    for (int j = 0; j < 8; ++j) t[8 + j] = (bf16_t)rp[kc + 8 + j];
    a[ks] = t;
  }

  v8f acc[4];
#pragma unroll
  for (int nt = 0; nt < 4; ++nt) {
    acc[nt] = vzero8();
#pragma unroll
    for (int ks = 0; ks < 2; ++ks) {
      // B[k=d][n=o] = W[o][d]: 16 contiguous d per lane-half
      v16bf b = *(const v16bf*)&wsm[(nt * 16 + lo) * DHEAD + ks * 32 + hi * 16];
      acc[nt] = wmma_bf16(a[ks], b, acc[nt]);
    }
  }

  if (which == 2) {
    // transposed store: Vt[n,h,d,s]; vgpr r runs over 8 consecutive s values
#pragma unroll
    for (int nt = 0; nt < 4; ++nt) {
      v8bf pk;
#pragma unroll
      for (int r = 0; r < 8; ++r) pk[r] = (bf16_t)acc[nt][r];
      bf16_t* op = Vt + ((size_t)(n * NHEADS + h) * DHEAD + nt * 16 + lo) * S_LEN
                      + s0 + wave * 16 + hi * 8;
      *(v8bf*)op = pk;
    }
  } else {
    bf16_t* outp = (which == 0) ? Qp : Kp;
#pragma unroll
    for (int nt = 0; nt < 4; ++nt) {
#pragma unroll
      for (int r = 0; r < 8; ++r) {
        const int srow = s0 + wave * 16 + hi * 8 + r;
        outp[((size_t)(n * NHEADS + h) * S_LEN + srow) * DHEAD + nt * 16 + lo] =
            (bf16_t)(acc[nt][r] * osc);
      }
    }
  }
}

// ---------------------------------------------------------------------------
// Kernel 2: flash attention, double-buffered K/V via async global->LDS DMA.
// Block = (n, h, 128 q rows); 8 waves x 16 rows.
// ---------------------------------------------------------------------------
__global__ __launch_bounds__(256) void attn_kernel(
    const bf16_t* __restrict__ Qp, const bf16_t* __restrict__ Kp,
    const bf16_t* __restrict__ Vt, const int* __restrict__ mask,
    bf16_t* __restrict__ AO)
{
  const int n  = blockIdx.z;
  const int h  = blockIdx.y;
  const int s0 = blockIdx.x * 128;
  const int tid  = threadIdx.x;
  const int wave = tid >> 5, lane = tid & 31;
  const int lo = lane & 15, hi = lane >> 4;

  __shared__ __align__(32) bf16_t kbuf[2 * 64 * 64];   // K tiles  [buf][kv][d]
  __shared__ __align__(32) bf16_t vbuf[2 * 64 * 64];   // V^T tiles [buf][d][kv]
  __shared__ __align__(32) bf16_t p_lds[8 * 16 * 64];  // per-wave P repack

  const bf16_t* kbase = Kp + (size_t)(n * NHEADS + h) * S_LEN * DHEAD;
  const bf16_t* vbase = Vt + (size_t)(n * NHEADS + h) * DHEAD * S_LEN;
  const unsigned long long kglob = (unsigned long long)(uintptr_t)kbase;
  const unsigned long long vglob = (unsigned long long)(uintptr_t)vbase;
  const uint32_t klds = (uint32_t)(uintptr_t)kbuf;  // flat low 32 = LDS offset
  const uint32_t vlds = (uint32_t)(uintptr_t)vbuf;

  // cooperative tile loader: 512 x 16B chunks; thread owns chunk tid & tid+256
  const uint32_t row0 = (uint32_t)(tid >> 3);        // 0..31
  const uint32_t sub  = (uint32_t)((tid & 7) * 16);  // byte offset within row

  auto issue_tile = [&](int kv0, int buf) {
    const uint32_t lk = klds + (uint32_t)buf * 8192 + row0 * 128 + sub;
    const uint32_t gk = (uint32_t)(kv0 * DHEAD * 2) + row0 * 128 + sub;
    async_b128(lk, gk, kglob);
    async_b128(lk + 4096, gk + 4096, kglob);  // rows 32..63
    const uint32_t lv = vlds + (uint32_t)buf * 8192 + row0 * 128 + sub;
    const uint32_t gv = row0 * (S_LEN * 2) + (uint32_t)(kv0 * 2) + sub;
    async_b128(lv, gv, vglob);
    async_b128(lv + 4096, gv + 32u * (S_LEN * 2), vglob);  // d rows 32..63
  };

  // Q A-fragments held in registers for the whole kernel
  const bf16_t* qrp =
      Qp + ((size_t)(n * NHEADS + h) * S_LEN + s0 + wave * 16 + lo) * DHEAD;
  v16bf aq[2];
#pragma unroll
  for (int ks = 0; ks < 2; ++ks) {
    const int kc = ks * 32 + hi * 16;
    v8bf alo = *(const v8bf*)(qrp + kc);
    v8bf ahi = *(const v8bf*)(qrp + kc + 8);
    aq[ks] = comb(alo, ahi);
  }

  v8f oacc[4];
  float mrow[8], lrow[8];
#pragma unroll
  for (int dt = 0; dt < 4; ++dt) oacc[dt] = vzero8();
#pragma unroll
  for (int r = 0; r < 8; ++r) { mrow[r] = -1e30f; lrow[r] = 0.0f; }

  const int* mbase = mask + (size_t)n * S_LEN * S_LEN;
  const int NT = S_LEN / 64;

  issue_tile(0, 0);  // prologue: tile 0 in flight

  for (int kt = 0; kt < NT; ++kt) {
    const int kv0 = kt * 64;
    const int buf = kt & 1;
    if (kt + 1 < NT) {
      issue_tile(kv0 + 64, buf ^ 1);  // overlap next tile's DMA with compute
      wait_async_le4();               // current tile's 4 ops done (in-order)
    } else {
      wait_async_le0();
    }
    __syncthreads();  // tile visible to all waves

    const bf16_t* kt_lds = kbuf + buf * 4096;
    const bf16_t* vt_lds = vbuf + buf * 4096;

    // S tile = Q @ K^T (4 column tiles of 16)
    v8f e[4];
#pragma unroll
    for (int nt = 0; nt < 4; ++nt) {
      e[nt] = vzero8();
#pragma unroll
      for (int ks = 0; ks < 2; ++ks) {
        v16bf b = *(const v16bf*)&kt_lds[(nt * 16 + lo) * 64 + ks * 32 + hi * 16];
        e[nt] = wmma_bf16(aq[ks], b, e[nt]);
      }
    }

    // mask (select, no branch -> EXEC stays all-1s); scale already in Q
#pragma unroll
    for (int nt = 0; nt < 4; ++nt) {
#pragma unroll
      for (int r = 0; r < 8; ++r) {
        const int qs = s0 + wave * 16 + hi * 8 + r;
        const int kc = kv0 + nt * 16 + lo;
        const int mv = mbase[(size_t)qs * S_LEN + kc];
        e[nt][r] = (mv == 0) ? -1e20f : e[nt][r];
      }
    }

    // online softmax; rows live in 16-lane halves -> xor-shuffle reductions
#pragma unroll
    for (int r = 0; r < 8; ++r) {
      float mx = fmaxf(fmaxf(e[0][r], e[1][r]), fmaxf(e[2][r], e[3][r]));
      mx = fmaxf(mx, __shfl_xor(mx, 1, 32));
      mx = fmaxf(mx, __shfl_xor(mx, 2, 32));
      mx = fmaxf(mx, __shfl_xor(mx, 4, 32));
      mx = fmaxf(mx, __shfl_xor(mx, 8, 32));
      const float mnew  = fmaxf(mrow[r], mx);
      const float alpha = __expf(mrow[r] - mnew);
      mrow[r] = mnew;
      float s = 0.0f;
#pragma unroll
      for (int nt = 0; nt < 4; ++nt) {
        const float p = __expf(e[nt][r] - mnew);
        e[nt][r] = p;
        s += p;
      }
      s += __shfl_xor(s, 1, 32);
      s += __shfl_xor(s, 2, 32);
      s += __shfl_xor(s, 4, 32);
      s += __shfl_xor(s, 8, 32);
      lrow[r] = lrow[r] * alpha + s;
#pragma unroll
      for (int dt = 0; dt < 4; ++dt) oacc[dt][r] *= alpha;
    }

    // repack P (C-layout f32) -> A-layout via per-wave LDS slab
#pragma unroll
    for (int nt = 0; nt < 4; ++nt)
#pragma unroll
      for (int r = 0; r < 8; ++r)
        p_lds[wave * 1024 + (hi * 8 + r) * 64 + nt * 16 + lo] = (bf16_t)e[nt][r];
    __syncthreads();

    // O += P @ V
#pragma unroll
    for (int ks = 0; ks < 2; ++ks) {
      const bf16_t* pp = &p_lds[wave * 1024 + lo * 64 + ks * 32 + hi * 16];
      v8bf plo = *(const v8bf*)pp;
      v8bf phi = *(const v8bf*)(pp + 8);
      v16bf ap = comb(plo, phi);
#pragma unroll
      for (int dt = 0; dt < 4; ++dt) {
        // B[k=kv][n=d] = V[kv][d] = vt_lds[d][kv], contiguous kv per lane-half
        v16bf b = *(const v16bf*)&vt_lds[(dt * 16 + lo) * 64 + ks * 32 + hi * 16];
        oacc[dt] = wmma_bf16(ap, b, oacc[dt]);
      }
    }
    __syncthreads();  // all waves done reading this buffer before its reuse
  }

  // normalize and store bf16 activations [n, s, E] for the output GEMM
#pragma unroll
  for (int dt = 0; dt < 4; ++dt) {
#pragma unroll
    for (int r = 0; r < 8; ++r) {
      const int srow = s0 + wave * 16 + hi * 8 + r;
      const float oo = oacc[dt][r] * (1.0f / lrow[r]);
      AO[((size_t)n * S_LEN + srow) * EMB + h * DHEAD + dt * 16 + lo] = (bf16_t)oo;
    }
  }
}

// ---------------------------------------------------------------------------
// Kernel 3: out = X @ Wo^T + bo   (4096 x 1024 x 1024), fp32 result
// ---------------------------------------------------------------------------
__global__ __launch_bounds__(256) void oproj_kernel(
    const bf16_t* __restrict__ X, const float* __restrict__ Wo,
    const float* __restrict__ bo, float* __restrict__ out)
{
  const int m0 = blockIdx.x * 128;
  const int o0 = blockIdx.y * 64;
  const int tid  = threadIdx.x;
  const int wave = tid >> 5, lane = tid & 31;
  const int lo = lane & 15, hi = lane >> 4;

  __shared__ __align__(32) bf16_t wsm[64 * 64];  // Wo tile [o_local][k_local]

  v8f acc[4];
#pragma unroll
  for (int nt = 0; nt < 4; ++nt) acc[nt] = vzero8();

  const bf16_t* rp = X + (size_t)(m0 + wave * 16 + lo) * EMB;

  for (int kt = 0; kt < EMB / 64; ++kt) {
    const int k0 = kt * 64;
    __syncthreads();
#pragma unroll 4
    for (int i = tid; i < 64 * 64; i += 256) {
      const int ol = i >> 6, kl = i & 63;
      wsm[i] = (bf16_t)Wo[(size_t)(o0 + ol) * EMB + k0 + kl];
      if (kt + 1 < EMB / 64)  // gfx1250 global_prefetch of next weight tile
        __builtin_prefetch(&Wo[(size_t)(o0 + ol) * EMB + k0 + 64 + kl], 0, 0);
    }
    __syncthreads();

#pragma unroll
    for (int ks = 0; ks < 2; ++ks) {
      const int kc = k0 + ks * 32 + hi * 16;
      v8bf alo = *(const v8bf*)(rp + kc);
      v8bf ahi = *(const v8bf*)(rp + kc + 8);
      v16bf a = comb(alo, ahi);
#pragma unroll
      for (int nt = 0; nt < 4; ++nt) {
        v16bf b = *(const v16bf*)&wsm[(nt * 16 + lo) * 64 + ks * 32 + hi * 16];
        acc[nt] = wmma_bf16(a, b, acc[nt]);
      }
    }
  }

#pragma unroll
  for (int nt = 0; nt < 4; ++nt) {
    const float bias = bo[o0 + nt * 16 + lo];
#pragma unroll
    for (int r = 0; r < 8; ++r) {
      out[(size_t)(m0 + wave * 16 + hi * 8 + r) * EMB + o0 + nt * 16 + lo] =
          acc[nt][r] + bias;
    }
  }
}

// ---------------------------------------------------------------------------
extern "C" void kernel_launch(void* const* d_in, const int* in_sizes, int n_in,
                              void* d_out, int out_size, void* d_ws, size_t ws_size,
                              hipStream_t stream) {
  (void)in_sizes; (void)n_in; (void)out_size; (void)ws_size;
  const float* values  = (const float*)d_in[0];
  const float* keys    = (const float*)d_in[1];
  const float* queries = (const float*)d_in[2];
  const int*   mask    = (const int*)d_in[3];
  const float* Wv = (const float*)d_in[4];
  const float* Wk = (const float*)d_in[5];
  const float* Wq = (const float*)d_in[6];
  const float* Wo = (const float*)d_in[7];
  const float* bo = (const float*)d_in[8];
  float* out = (float*)d_out;

  const size_t per = (size_t)NBATCH * NHEADS * S_LEN * DHEAD;  // 4M bf16 elems
  bf16_t* Qp = (bf16_t*)d_ws;
  bf16_t* Kp = Qp + per;
  bf16_t* Vt = Kp + per;
  bf16_t* AO = Vt + per;  // [N, S, E] bf16

  proj_kernel<<<dim3(S_LEN / 128, NHEADS, NBATCH * 3), 256, 0, stream>>>(
      queries, keys, values, Wq, Wk, Wv, Qp, Kp, Vt);
  attn_kernel<<<dim3(S_LEN / 128, NHEADS, NBATCH), 256, 0, stream>>>(
      Qp, Kp, Vt, mask, AO);
  oproj_kernel<<<dim3((NBATCH * S_LEN) / 128, EMB / 64), 256, 0, stream>>>(
      AO, Wo, bo, out);
}
